// AlexCapsNet_Fully_FOOD101_87488483820211
// MI455X (gfx1250) — matrix operand order repaired and checked by reading, not compiled, not run on hardware
//
#include <hip/hip_runtime.h>
#include <hip/hip_bf16.h>
#include <math.h>

// ---------------- bf16 helpers (bit-exact storage as ushort) ----------------
typedef __attribute__((ext_vector_type(16))) __bf16 v16bf;
typedef __attribute__((ext_vector_type(8)))  float  v8f;

__device__ __forceinline__ unsigned short f32_to_bf16_bits(float f) {
  union { float f; unsigned u; } x; x.f = f;
  unsigned u = x.u;
  u += 0x7FFFu + ((u >> 16) & 1u);      // round-to-nearest-even
  return (unsigned short)(u >> 16);
}
__device__ __forceinline__ float bf16_bits_to_f32(unsigned short h) {
  union { unsigned u; float f; } x; x.u = ((unsigned)h) << 16;
  return x.f;
}
__device__ __forceinline__ __bf16 bf16_from_bits(unsigned short h) {
  union { unsigned short u; __bf16 b; } x; x.u = h;
  return x.b;
}

// ---------------- layout / cast kernels ----------------
__global__ void nchw_to_nhwc_bf16(const float* __restrict__ in, unsigned short* __restrict__ out,
                                  int B, int C, int H, int W) {
  long long idx = (long long)blockIdx.x * blockDim.x + threadIdx.x;
  long long total = (long long)B * C * H * W;
  if (idx >= total) return;
  int c = idx % C; long long t = idx / C;
  int x = t % W; t /= W;
  int y = t % H; int b = (int)(t / H);
  float v = in[(((long long)b * C + c) * H + y) * W + x];
  out[idx] = f32_to_bf16_bits(v);   // idx already == ((b*H+y)*W+x)*C + c
}

// OIHW f32 -> [K = KH*KW*Cin, N = Cout] bf16, k = (ky*KW+kx)*Cin + c
__global__ void weight_to_kn_bf16(const float* __restrict__ w, unsigned short* __restrict__ out,
                                  int Cout, int Cin, int KH, int KW) {
  long long idx = (long long)blockIdx.x * blockDim.x + threadIdx.x;
  long long total = (long long)Cout * Cin * KH * KW;
  if (idx >= total) return;
  int n = idx % Cout;
  long long k = idx / Cout;
  int c = k % Cin; long long t = k / Cin;
  int kx = t % KW; int ky = (int)(t / KW);
  float v = w[(((long long)n * Cin + c) * KH + ky) * KW + kx];
  out[idx] = f32_to_bf16_bits(v);
}

__global__ void cast_f32_to_bf16(const float* __restrict__ in, unsigned short* __restrict__ out,
                                 long long n) {
  long long idx = (long long)blockIdx.x * blockDim.x + threadIdx.x;
  if (idx < n) out[idx] = f32_to_bf16_bits(in[idx]);
}

__global__ void fill_zero_f32(float* __restrict__ p, long long n) {
  long long idx = (long long)blockIdx.x * blockDim.x + threadIdx.x;
  if (idx < n) p[idx] = 0.0f;
}

// ---------------- maxpool (NHWC bf16, -inf padding semantics) ----------------
__global__ void maxpool_nhwc_bf16(const unsigned short* __restrict__ in, unsigned short* __restrict__ out,
                                  int B, int H, int W, int C, int KS, int S, int P, int OH, int OW) {
  long long idx = (long long)blockIdx.x * blockDim.x + threadIdx.x;
  long long total = (long long)B * OH * OW * C;
  if (idx >= total) return;
  int c = idx % C; long long t = idx / C;
  int ox = t % OW; t /= OW;
  int oy = t % OH; int b = (int)(t / OH);
  float m = -3.402823466e38f;
  for (int ky = 0; ky < KS; ++ky) {
    int iy = oy * S + ky - P;
    if (iy < 0 || iy >= H) continue;
    for (int kx = 0; kx < KS; ++kx) {
      int ix = ox * S + kx - P;
      if (ix < 0 || ix >= W) continue;
      float v = bf16_bits_to_f32(in[(((long long)b * H + iy) * W + ix) * C + c]);
      m = fmaxf(m, v);
    }
  }
  out[idx] = f32_to_bf16_bits(m);
}

// ---------------- implicit-im2col GEMM with WMMA bf16 (fully templated) ----------------
// D[M=BN*OH*OW, N=COUT] = im2col(A_src NHWC bf16) x Bw[K,N] bf16 + bias
// Block tile 64x64, 8 waves: wave grid 4(M) x 2(N), each wave 16x32 C -> 2 WMMA/K-step.
// Register double-buffer pipeline: next tile's global loads overlap current WMMAs.
template <bool RELU, int BN, int H, int W, int CIN, int KH, int KW,
          int STRIDE, int PAD, int OH, int OW, int COUT>
__global__ void __launch_bounds__(256)
conv_gemm_wmma(const unsigned short* __restrict__ A_src,
               const unsigned short* __restrict__ Bw,
               const float* __restrict__ bias,
               unsigned short* __restrict__ out_bf,   // may be null
               float* __restrict__ out_f32) {         // may be null
  constexpr int M = BN * OH * OW;
  constexpr int N = COUT;
  constexpr int K = KH * KW * CIN;
  constexpr int KSTEPS = (K + 31) / 32;
  constexpr bool FAST_A = (CIN % 32) == 0;         // 32-wide K-slice within one (ky,kx) plane
  constexpr bool A_1X1  = (KH == 1 && KW == 1);    // contiguous feature row
  constexpr bool FAST_B = (COUT % 8) == 0;

  __shared__ unsigned short As[64][32];
  __shared__ unsigned short Bs[32][64];

  const int tid  = threadIdx.x;
  const int lane = tid & 31;
  const int wave = tid >> 5;
  const int waveM = wave >> 1;          // 0..3
  const int waveN = wave & 1;           // 0..1
  const int blockM = blockIdx.y;
  const int blockN = blockIdx.x;
  const int row0 = blockM * 64 + waveM * 16;
  const int col0 = blockN * 64 + waveN * 32;

  // A-tile: 64 rows x 32 K, 8 halves/thread
  const int arow  = tid >> 2;            // 0..63
  const int acol0 = (tid & 3) * 8;       // 0,8,16,24
  const int m_a = blockM * 64 + arow;
  const bool mval = (m_a < M);
  int n_pix = 0, oy = 0, ox = 0;
  if (mval) {
    n_pix = m_a / (OH * OW);             // constant divisors -> mul/shift
    const int rem = m_a % (OH * OW);
    oy = rem / OW; ox = rem % OW;
  }
  // B-tile: 32 K x 64 N, 8 halves/thread
  const int brow  = tid >> 3;            // 0..31
  const int bcol0 = (tid & 7) * 8;       // 0..56
  const int bcol  = blockN * 64 + bcol0;

  union Pack { uint4 v; unsigned short s[8]; };

  auto loadA = [&](int kk) {
    Pack a; a.v = make_uint4(0u, 0u, 0u, 0u);
    const int k0 = kk * 32;
    if constexpr (FAST_A) {
      const int plane = k0 / CIN;                  // uniform per K-step
      const int cbase = k0 - plane * CIN + acol0;  // contiguous 8 channels
      const int kx = plane % KW;
      const int ky = plane / KW;
      const int iy = oy * STRIDE + ky - PAD;
      const int ix = ox * STRIDE + kx - PAD;
      if (mval && iy >= 0 && iy < H && ix >= 0 && ix < W)
        a.v = *(const uint4*)(A_src + (((long long)n_pix * H + iy) * W + ix) * CIN + cbase);
    } else if constexpr (A_1X1) {
      const int cbase = k0 + acol0;
      if (mval) {
        if (cbase + 8 <= K) {
          a.v = *(const uint4*)(A_src + (long long)n_pix * K + cbase);
        } else {
#pragma unroll
          for (int q = 0; q < 8; ++q)
            if (cbase + q < K) a.s[q] = A_src[(long long)n_pix * K + cbase + q];
        }
      }
    } else {
#pragma unroll
      for (int q = 0; q < 8; ++q) {
        const int k = k0 + acol0 + q;
        if (mval && k < K) {
          const int c = k % CIN;                   // constant divisors (e.g. CIN=3)
          const int t = k / CIN;
          const int kx = t % KW;
          const int ky = t / KW;
          const int iy = oy * STRIDE + ky - PAD;
          const int ix = ox * STRIDE + kx - PAD;
          if (iy >= 0 && iy < H && ix >= 0 && ix < W)
            a.s[q] = A_src[(((long long)n_pix * H + iy) * W + ix) * CIN + c];
        }
      }
    }
    return a;
  };

  auto loadB = [&](int kk) {
    Pack b; b.v = make_uint4(0u, 0u, 0u, 0u);
    const int k = kk * 32 + brow;
    if constexpr (FAST_B) {
      if (k < K && bcol < N)
        b.v = *(const uint4*)(Bw + (long long)k * N + bcol);
    } else {
#pragma unroll
      for (int q = 0; q < 8; ++q)
        if (k < K && bcol + q < N) b.s[q] = Bw[(long long)k * N + bcol + q];
    }
    return b;
  };

  v8f acc0 = {};
  v8f acc1 = {};
  Pack aCur = loadA(0);
  Pack bCur = loadB(0);

  for (int kk = 0; kk < KSTEPS; ++kk) {
    *(uint4*)&As[arow][acol0] = aCur.v;            // ds_store_b128
    *(uint4*)&Bs[brow][bcol0] = bCur.v;
    __syncthreads();

    Pack aNext, bNext;
    aNext.v = make_uint4(0u, 0u, 0u, 0u);
    bNext.v = make_uint4(0u, 0u, 0u, 0u);
    if (kk + 1 < KSTEPS) { aNext = loadA(kk + 1); bNext = loadB(kk + 1); }

    // fragments per ISA 7.12.2 bf16 layouts
    v16bf afrag, bfrag0, bfrag1;
    const int r  = lane & 15;
    const int kb = (lane >> 4) * 8;
#pragma unroll
    for (int h = 0; h < 8; ++h) {
      afrag[h]     = bf16_from_bits(As[waveM * 16 + r][kb + h]);
      afrag[h + 8] = bf16_from_bits(As[waveM * 16 + r][kb + 16 + h]);
    }
#pragma unroll
    for (int h = 0; h < 16; ++h) {
      bfrag0[h] = bf16_from_bits(Bs[kb + h + (h >= 8 ? 8 : 0)][waveN * 32 + r]);
      bfrag1[h] = bf16_from_bits(Bs[kb + h + (h >= 8 ? 8 : 0)][waveN * 32 + 16 + r]);
    }
    acc0 = __builtin_amdgcn_wmma_f32_16x16x32_bf16(false, afrag, false, bfrag0,
                                                   (short)0, acc0, false, false);
    acc1 = __builtin_amdgcn_wmma_f32_16x16x32_bf16(false, afrag, false, bfrag1,
                                                   (short)0, acc1, false, false);
    __syncthreads();
    aCur = aNext; bCur = bNext;
  }

  // epilogue: C layout lanes0-15 {M=v,N=lane}, lanes16-31 {M=v+8,N=lane-16}
  const int nOut0 = col0 + (lane & 15);
  const int nOut1 = nOut0 + 16;
  const int mBase = row0 + ((lane >> 4) * 8);
#pragma unroll
  for (int v = 0; v < 8; ++v) {
    const int m = mBase + v;
    if (m < M) {
      if (nOut0 < N) {
        float val = acc0[v] + bias[nOut0];
        if (RELU) val = fmaxf(val, 0.0f);
        if (out_bf)  out_bf[(long long)m * N + nOut0]  = f32_to_bf16_bits(val);
        if (out_f32) out_f32[(long long)m * N + nOut0] = val;
      }
      if (nOut1 < N) {
        float val = acc1[v] + bias[nOut1];
        if (RELU) val = fmaxf(val, 0.0f);
        if (out_bf)  out_bf[(long long)m * N + nOut1]  = f32_to_bf16_bits(val);
        if (out_f32) out_f32[(long long)m * N + nOut1] = val;
      }
    }
  }
}

// ---------------- PrimaryCaps gather (NCHW reshape semantics) + squash ----------------
__global__ void primary_squash(const float* __restrict__ p, float* __restrict__ u, int B) {
  int t = blockIdx.x * blockDim.x + threadIdx.x;
  if (t >= B * 1152) return;
  int b = t / 1152, i = t % 1152;
  float s[8]; float sq = 0.0f;
#pragma unroll
  for (int d = 0; d < 8; ++d) {
    int idx = i * 8 + d;
    int c = idx / 36, rem = idx % 36;
    int y = rem / 6, x = rem % 6;
    float v = p[(((long long)b * 6 + y) * 6 + x) * 256 + c];
    s[d] = v; sq += v * v;
  }
  float scale = (sq / (1.0f + sq)) / sqrtf(sq + 1e-8f);
#pragma unroll
  for (int d = 0; d < 8; ++d) u[((long long)b * 1152 + i) * 8 + d] = s[d] * scale;
}

// ---------------- routing: softmax over 101 out-caps ----------------
__global__ void routing_softmax(const float* __restrict__ bl, float* __restrict__ cc, int B) {
  int t = blockIdx.x * blockDim.x + threadIdx.x;
  if (t >= B * 1152) return;
  int b = t / 1152, i = t % 1152;
  long long base = (long long)b * 101 * 1152 + i;
  float mx = -3.402823466e38f;
  for (int j = 0; j < 101; ++j) mx = fmaxf(mx, bl[base + (long long)j * 1152]);
  float sum = 0.0f;
  for (int j = 0; j < 101; ++j) {
    float e = __expf(bl[base + (long long)j * 1152] - mx);
    cc[base + (long long)j * 1152] = e;
    sum += e;
  }
  float inv = 1.0f / sum;
  for (int j = 0; j < 101; ++j) cc[base + (long long)j * 1152] *= inv;
}

// ---------------- routing: v[b,j,o] = squash(sum_i c[b,j,i] * xhat), xhat recomputed ----------------
__global__ void __launch_bounds__(128)
routing_v(const float* __restrict__ C, const float* __restrict__ U,
          const float* __restrict__ Wc, float* __restrict__ V) {
  const int bj = blockIdx.x;
  const int b = bj / 101, j = bj % 101;
  const int tid = threadIdx.x;
  float acc[16];
#pragma unroll
  for (int o = 0; o < 16; ++o) acc[o] = 0.0f;
  for (int i = tid; i < 1152; i += 128) {
    const float ci = C[((long long)b * 101 + j) * 1152 + i];
    const float* ur = U + ((long long)b * 1152 + i) * 8;
    const float* wr = Wc + ((long long)j * 1152 + i) * 128;   // 16*8
    float uu[8];
#pragma unroll
    for (int d = 0; d < 8; ++d) uu[d] = ur[d];
#pragma unroll
    for (int o = 0; o < 16; ++o) {
      float xh = 0.0f;
#pragma unroll
      for (int d = 0; d < 8; ++d) xh += wr[o * 8 + d] * uu[d];
      acc[o] += ci * xh;
    }
  }
  __shared__ float red[128][16];
#pragma unroll
  for (int o = 0; o < 16; ++o) red[tid][o] = acc[o];
  __syncthreads();
  for (int s = 64; s > 0; s >>= 1) {
    if (tid < s)
      for (int o = 0; o < 16; ++o) red[tid][o] += red[tid + s][o];
    __syncthreads();
  }
  if (tid == 0) {
    float sq = 0.0f;
    for (int o = 0; o < 16; ++o) sq += red[0][o] * red[0][o];
    float scale = (sq / (1.0f + sq)) / sqrtf(sq + 1e-8f);
    for (int o = 0; o < 16; ++o)
      V[((long long)b * 101 + j) * 16 + o] = red[0][o] * scale;
  }
}

// ---------------- routing: b_logits += <v[b,j], xhat[b,j,i]> ----------------
__global__ void routing_update(const float* __restrict__ U, const float* __restrict__ Wc,
                               const float* __restrict__ V, float* __restrict__ bl, int B) {
  long long t = (long long)blockIdx.x * blockDim.x + threadIdx.x;
  long long total = (long long)B * 101 * 1152;
  if (t >= total) return;
  int i = t % 1152; long long t2 = t / 1152;
  int j = t2 % 101; int b = (int)(t2 / 101);
  const float* ur = U + ((long long)b * 1152 + i) * 8;
  const float* wr = Wc + ((long long)j * 1152 + i) * 128;
  const float* vr = V + ((long long)b * 101 + j) * 16;
  float uu[8];
#pragma unroll
  for (int d = 0; d < 8; ++d) uu[d] = ur[d];
  float dot = 0.0f;
#pragma unroll
  for (int o = 0; o < 16; ++o) {
    float xh = 0.0f;
#pragma unroll
    for (int d = 0; d < 8; ++d) xh += wr[o * 8 + d] * uu[d];
    dot += vr[o] * xh;
  }
  bl[t] += dot;
}

// ---------------- host side ----------------
static inline long long cdiv_ll(long long a, long long b) { return (a + b - 1) / b; }

extern "C" void kernel_launch(void* const* d_in, const int* in_sizes, int n_in,
                              void* d_out, int out_size, void* d_ws, size_t ws_size,
                              hipStream_t stream) {
  const float* x     = (const float*)d_in[0];
  const float* w1    = (const float*)d_in[1];  const float* b1 = (const float*)d_in[2];
  const float* w2    = (const float*)d_in[3];  const float* b2 = (const float*)d_in[4];
  const float* w3    = (const float*)d_in[5];  const float* b3 = (const float*)d_in[6];
  const float* w4    = (const float*)d_in[7];  const float* b4 = (const float*)d_in[8];
  const float* w5    = (const float*)d_in[9];  const float* b5 = (const float*)d_in[10];
  const float* pc_w  = (const float*)d_in[11]; const float* pc_b = (const float*)d_in[12];
  const float* capsW = (const float*)d_in[13];
  const float* f1w   = (const float*)d_in[14]; const float* f1b = (const float*)d_in[15];
  const float* f2w   = (const float*)d_in[16]; const float* f2b = (const float*)d_in[17];
  const float* f3w   = (const float*)d_in[18]; const float* f3b = (const float*)d_in[19];
  float* out = (float*)d_out;

  const int B = 64;
  char* wsp = (char*)d_ws;
  size_t off = 0;
  auto alloc = [&](size_t bytes) -> void* {
    void* p = wsp + off;
    off += (bytes + 255) & ~(size_t)255;
    return p;
  };

  unsigned short* x_nhwc = (unsigned short*)alloc((size_t)B * 227 * 227 * 3 * 2);
  unsigned short* wt1 = (unsigned short*)alloc((size_t)363 * 96 * 2);
  unsigned short* wt2 = (unsigned short*)alloc((size_t)2400 * 256 * 2);
  unsigned short* wt3 = (unsigned short*)alloc((size_t)2304 * 384 * 2);
  unsigned short* wt4 = (unsigned short*)alloc((size_t)3456 * 384 * 2);
  unsigned short* wt5 = (unsigned short*)alloc((size_t)3456 * 256 * 2);
  unsigned short* wtp = (unsigned short*)alloc((size_t)2304 * 256 * 2);
  unsigned short* wtf1 = (unsigned short*)alloc((size_t)1616 * 4096 * 2);
  unsigned short* wtf2 = (unsigned short*)alloc((size_t)4096 * 4096 * 2);
  unsigned short* wtf3 = (unsigned short*)alloc((size_t)4096 * 101 * 2);
  unsigned short* a1 = (unsigned short*)alloc((size_t)B * 55 * 55 * 96 * 2);
  unsigned short* p1 = (unsigned short*)alloc((size_t)B * 27 * 27 * 96 * 2);
  unsigned short* a2 = (unsigned short*)alloc((size_t)B * 27 * 27 * 256 * 2);
  unsigned short* p2 = (unsigned short*)alloc((size_t)B * 14 * 14 * 256 * 2);
  unsigned short* a3 = (unsigned short*)alloc((size_t)B * 14 * 14 * 384 * 2);
  unsigned short* a4 = (unsigned short*)alloc((size_t)B * 14 * 14 * 384 * 2);
  unsigned short* a5 = (unsigned short*)alloc((size_t)B * 14 * 14 * 256 * 2);
  unsigned short* p3 = (unsigned short*)alloc((size_t)B * 6 * 6 * 256 * 2);
  float* pc_out = (float*)alloc((size_t)B * 6 * 6 * 256 * 4);
  float* u      = (float*)alloc((size_t)B * 1152 * 8 * 4);
  float* bl     = (float*)alloc((size_t)B * 101 * 1152 * 4);
  float* cc     = (float*)alloc((size_t)B * 101 * 1152 * 4);
  float* vcaps  = (float*)alloc((size_t)B * 101 * 16 * 4);
  unsigned short* feat = (unsigned short*)alloc((size_t)B * 1616 * 2);
  unsigned short* fa1  = (unsigned short*)alloc((size_t)B * 4096 * 2);
  unsigned short* fa2  = (unsigned short*)alloc((size_t)B * 4096 * 2);
  (void)ws_size; (void)in_sizes; (void)n_in; (void)out_size;

  const int T = 256;
  auto g1 = [&](long long n) { return dim3((unsigned)cdiv_ll(n, T)); };

  // input + weight prep
  nchw_to_nhwc_bf16<<<g1((long long)B * 3 * 227 * 227), T, 0, stream>>>(x, x_nhwc, B, 3, 227, 227);
  weight_to_kn_bf16<<<g1(363LL * 96), T, 0, stream>>>(w1, wt1, 96, 3, 11, 11);
  weight_to_kn_bf16<<<g1(2400LL * 256), T, 0, stream>>>(w2, wt2, 256, 96, 5, 5);
  weight_to_kn_bf16<<<g1(2304LL * 384), T, 0, stream>>>(w3, wt3, 384, 256, 3, 3);
  weight_to_kn_bf16<<<g1(3456LL * 384), T, 0, stream>>>(w4, wt4, 384, 384, 3, 3);
  weight_to_kn_bf16<<<g1(3456LL * 256), T, 0, stream>>>(w5, wt5, 256, 384, 3, 3);
  weight_to_kn_bf16<<<g1(2304LL * 256), T, 0, stream>>>(pc_w, wtp, 256, 256, 3, 3);
  weight_to_kn_bf16<<<g1(1616LL * 4096), T, 0, stream>>>(f1w, wtf1, 4096, 1616, 1, 1);
  weight_to_kn_bf16<<<g1(4096LL * 4096), T, 0, stream>>>(f2w, wtf2, 4096, 4096, 1, 1);
  weight_to_kn_bf16<<<g1(4096LL * 101), T, 0, stream>>>(f3w, wtf3, 101, 4096, 1, 1);

  // conv stack: grid = (ceil(N/64), ceil(M/64))
  conv_gemm_wmma<true, 64, 227, 227, 3, 11, 11, 4, 0, 55, 55, 96>
      <<<dim3(2, 3025), 256, 0, stream>>>(x_nhwc, wt1, b1, a1, nullptr);
  maxpool_nhwc_bf16<<<g1((long long)B * 27 * 27 * 96), T, 0, stream>>>(a1, p1, B, 55, 55, 96, 3, 2, 0, 27, 27);
  conv_gemm_wmma<true, 64, 27, 27, 96, 5, 5, 1, 2, 27, 27, 256>
      <<<dim3(4, 729), 256, 0, stream>>>(p1, wt2, b2, a2, nullptr);
  maxpool_nhwc_bf16<<<g1((long long)B * 14 * 14 * 256), T, 0, stream>>>(a2, p2, B, 27, 27, 256, 3, 2, 1, 14, 14);
  conv_gemm_wmma<true, 64, 14, 14, 256, 3, 3, 1, 1, 14, 14, 384>
      <<<dim3(6, 196), 256, 0, stream>>>(p2, wt3, b3, a3, nullptr);
  conv_gemm_wmma<true, 64, 14, 14, 384, 3, 3, 1, 1, 14, 14, 384>
      <<<dim3(6, 196), 256, 0, stream>>>(a3, wt4, b4, a4, nullptr);
  conv_gemm_wmma<true, 64, 14, 14, 384, 3, 3, 1, 1, 14, 14, 256>
      <<<dim3(4, 196), 256, 0, stream>>>(a4, wt5, b5, a5, nullptr);
  maxpool_nhwc_bf16<<<g1((long long)B * 6 * 6 * 256), T, 0, stream>>>(a5, p3, B, 14, 14, 256, 3, 2, 0, 6, 6);

  // PrimaryCaps conv (f32 out, no relu) + squash
  conv_gemm_wmma<false, 64, 6, 6, 256, 3, 3, 1, 1, 6, 6, 256>
      <<<dim3(4, 36), 256, 0, stream>>>(p3, wtp, pc_b, nullptr, pc_out);
  primary_squash<<<g1((long long)B * 1152), T, 0, stream>>>(pc_out, u, B);

  // dynamic routing (3 iterations), x_hat recomputed from caps_W (L2-resident)
  const long long BL = (long long)B * 101 * 1152;
  fill_zero_f32<<<g1(BL), T, 0, stream>>>(bl, BL);
  for (int r = 0; r < 3; ++r) {
    routing_softmax<<<g1((long long)B * 1152), T, 0, stream>>>(bl, cc, B);
    routing_v<<<dim3(B * 101), 128, 0, stream>>>(cc, u, capsW, vcaps);
    if (r < 2)
      routing_update<<<g1(BL), T, 0, stream>>>(u, capsW, vcaps, bl, B);
  }

  // MLP head via the same WMMA GEMM (1x1 "conv")
  cast_f32_to_bf16<<<g1((long long)B * 1616), T, 0, stream>>>(vcaps, feat, (long long)B * 1616);
  conv_gemm_wmma<true, 64, 1, 1, 1616, 1, 1, 1, 0, 1, 1, 4096>
      <<<dim3(64, 1), 256, 0, stream>>>(feat, wtf1, f1b, fa1, nullptr);
  conv_gemm_wmma<true, 64, 1, 1, 4096, 1, 1, 1, 0, 1, 1, 4096>
      <<<dim3(64, 1), 256, 0, stream>>>(fa1, wtf2, f2b, fa2, nullptr);
  conv_gemm_wmma<false, 64, 1, 1, 4096, 1, 1, 1, 0, 1, 1, 101>
      <<<dim3(2, 1), 256, 0, stream>>>(fa2, wtf3, f3b, nullptr, out);
}
